// Transformer_psf_52853867544786
// MI455X (gfx1250) — matrix operand order repaired and checked by reading, not compiled, not run on hardware
//
#include <hip/hip_runtime.h>
#include <hip/hip_bf16.h>

// MI455X / gfx1250 fused implementation.
//
//  Kernel 1: 1x1 conv (f16 WMMA, f32 accum) + LayerNorm(W) fully in registers.
//  Kernel 2: 81-lag shifted correlation over H, pure VALU FMA with rolling
//            q-window registers and LDS-staged k rows (halo = implicit pad).
//
// Traffic ~1.4 GB -> ~60us at 23.3 TB/s; compute (34 GF WMMA + 11 GF VALU)
// sits well under that, so the kernels are organized for streaming.
//
// Round-2 tuning: LDS row stride 136 halves (272 B, 16B-aligned rows, 68-dword
// stride -> conflict-free b128 lane access) so A/B WMMA fragments load as
// ds_load_b128 instead of ds_load_2addr_b32 chains.

typedef __attribute__((ext_vector_type(16))) _Float16 v16h;
typedef __attribute__((ext_vector_type(8)))  _Float16 v8h;
typedef __attribute__((ext_vector_type(8)))  float    v8f;

#define BB 8
#define CC 128
#define DD 128
#define HH 512
#define WW 128
#define LDSP 136   // halves per LDS row: 272 B (16B aligned), 68 dwords

// ---------------------------------------------------------------------------
// Kernel 1: per (b,h) slab. D = Wm[128x128] * X[128x128] (contract over C),
// then LayerNorm over W per d-row, write to scratch.  Two passes: K then Q.
// ---------------------------------------------------------------------------
__global__ __launch_bounds__(256)
void conv_ln_kernel(const float* __restrict__ x,
                    const float* __restrict__ Wk,
                    const float* __restrict__ gk, const float* __restrict__ bkln,
                    const float* __restrict__ Wq,
                    const float* __restrict__ gq, const float* __restrict__ bqln,
                    float* __restrict__ kbuf, float* __restrict__ qbuf)
{
    __shared__ alignas(16) _Float16 sXT[WW][LDSP];  // [w][c] (B source, transposed)
    __shared__ alignas(16) _Float16 sW [DD][LDSP];  // [d][c] (A source)

    const int bh   = blockIdx.x;
    const int b    = bh >> 9;          // H = 512
    const int h    = bh & 511;
    const int tid  = threadIdx.x;
    const int lane = tid & 31;
    const int wv   = tid >> 5;         // 8 waves -> d-tile row
    const int m    = lane & 15;
    const int hi   = (lane >> 4) & 1;  // lane half: rows r (hi=0) / r+8 (hi=1)

    // ---- stage x[b, :, h, :] as f16, transposed to [w][c] ----
    const float* xp = x + ((size_t)b * CC * HH + h) * WW;
    for (int r = 0; r < 64; ++r) {
        int idx = r * 256 + tid;
        int c = idx >> 7, w = idx & 127;                 // w contiguous per tid
        sXT[w][c] = (_Float16)xp[(size_t)c * HH * WW + w];
    }

    for (int pass = 0; pass < 2; ++pass) {
        const float* Wm    = pass ? Wq   : Wk;
        const float* gamma = pass ? gq   : gk;
        const float* beta  = pass ? bqln : bkln;
        float*       outp  = pass ? qbuf : kbuf;

        __syncthreads();                                 // sW free to overwrite
        for (int r = 0; r < 64; ++r) {
            int idx = r * 256 + tid;
            sW[idx >> 7][idx & 127] = (_Float16)Wm[idx];
        }
        __syncthreads();

        // ---- A fragments (Wm rows, this wave's 16 d values), 4 K-chunks ----
        // A 16x32 layout (matches B): element i -> half [cc*32 + hi*16 + i]
        // i.e. one contiguous 32-byte run -> 2x ds_load_b128.
        v16h afr[4];
#pragma unroll
        for (int cc = 0; cc < 4; ++cc) {
            const _Float16* arow = &sW[wv * 16 + m][cc * 32 + hi * 16];
            v8h a0 = *(const v8h*)(arow);
            v8h a1 = *(const v8h*)(arow + 8);
            afr[cc] = __builtin_shufflevector(a0, a1, 0, 1, 2, 3, 4, 5, 6, 7,
                                              8, 9, 10, 11, 12, 13, 14, 15);
        }

        // ---- WMMA over 8 w-tiles, keep every accumulator in registers ----
        float accs[8][8];
#pragma unroll
        for (int wt = 0; wt < 8; ++wt) {
            v8f acc = {0.f, 0.f, 0.f, 0.f, 0.f, 0.f, 0.f, 0.f};
#pragma unroll
            for (int cc = 0; cc < 4; ++cc) {
                // B 32x16 layout: element i -> half [cc*32 + hi*16 + i] of row
                // n = wt*16 + (lane&15): one contiguous 32-byte run.
                v16h bfr = *(const v16h*)(&sXT[wt * 16 + m][cc * 32 + hi * 16]);
                acc = __builtin_amdgcn_wmma_f32_16x16x32_f16(
                        false, afr[cc], false, bfr, (short)0, acc, false, false);
            }
#pragma unroll
            for (int r = 0; r < 8; ++r) accs[wt][r] = acc[r];
        }

        // ---- LayerNorm over w, in registers. Row d = wv*16 + r + 8*hi owns
        //      w = wt*16 + (lane&15): per-lane partial over wt, then 16-lane
        //      butterfly (masks 1..8 stay inside each lane half). Conv bias is
        //      constant along w -> cancelled exactly by LN, so it is skipped.
        float s1[8], s2[8];
#pragma unroll
        for (int r = 0; r < 8; ++r) {
            float a = 0.f, q2 = 0.f;
#pragma unroll
            for (int wt = 0; wt < 8; ++wt) { float v = accs[wt][r]; a += v; q2 += v * v; }
            s1[r] = a; s2[r] = q2;
        }
#pragma unroll
        for (int mask = 1; mask < 16; mask <<= 1) {
#pragma unroll
            for (int r = 0; r < 8; ++r) {
                s1[r] += __shfl_xor(s1[r], mask, 32);
                s2[r] += __shfl_xor(s2[r], mask, 32);
            }
        }
        float mean[8], rstd[8];
#pragma unroll
        for (int r = 0; r < 8; ++r) {
            mean[r] = s1[r] * (1.0f / 128.0f);
            float var = s2[r] * (1.0f / 128.0f) - mean[r] * mean[r];
            rstd[r] = rsqrtf(var + 1e-5f);
        }

        // ---- normalized store: out[b, d, h, w] ----
        size_t obase = (size_t)b * DD * HH * WW + (size_t)h * WW;
#pragma unroll
        for (int wt = 0; wt < 8; ++wt) {
            int w = wt * 16 + m;
            float g = gamma[w], be = beta[w];
#pragma unroll
            for (int r = 0; r < 8; ++r) {
                int d = wv * 16 + r + 8 * hi;
                outp[obase + (size_t)d * HH * WW + w] =
                    (accs[wt][r] - mean[r]) * rstd[r] * g + be;
            }
        }
        __syncthreads();                                 // LDS reuse barrier
    }
}

// ---------------------------------------------------------------------------
// Kernel 2: per (b,d) channel plane.  atten[di,dj][w] = sum_h q[h,w]*k[h+di-4,
// w+dj-4].  Loop over k rows r: rolling 9-register q window (q rows r+4..r-4),
// k row staged in double-buffered LDS with zero halo (implicit jnp.pad).
// 81 all-register FMAs per row per lane; 9 LDS reads per row (DS:VALU = 1:9).
// ---------------------------------------------------------------------------
__global__ __launch_bounds__(128)
void corr_kernel(const float* __restrict__ qb, const float* __restrict__ kb,
                 const float* __restrict__ rpb, float* __restrict__ out)
{
    __shared__ float sK[2][136];                 // 4 halo + 128 + 4 halo
    const int bd  = blockIdx.x;                  // b*128 + d
    const int tid = threadIdx.x;                 // = w
    const size_t base = (size_t)bd * (HH * WW);

    if (tid < 4) {
        sK[0][tid] = 0.f; sK[1][tid] = 0.f;
        sK[0][132 + tid] = 0.f; sK[1][132 + tid] = 0.f;
    }

    float acc[81];
#pragma unroll
    for (int i = 0; i < 81; ++i) acc[i] = 0.f;

    float qwin[9];                               // qwin[di] = q[r+4-di, w]
#pragma unroll
    for (int i = 0; i < 9; ++i) qwin[i] = 0.f;
#pragma unroll
    for (int rr = 0; rr < 5; ++rr)
        qwin[4 - rr] = qb[base + (size_t)rr * WW + tid];
    __syncthreads();

    for (int r = 0; r < HH; ++r) {
        float* kr = sK[r & 1];
        kr[4 + tid] = kb[base + (size_t)r * WW + tid];
        float qnext = (r + 5 < HH) ? qb[base + (size_t)(r + 5) * WW + tid] : 0.f;
        __syncthreads();

        float kv[9];
#pragma unroll
        for (int dj = 0; dj < 9; ++dj) kv[dj] = kr[tid + dj];   // w + dj - 4

#pragma unroll
        for (int di = 0; di < 9; ++di)
#pragma unroll
            for (int dj = 0; dj < 9; ++dj)
                acc[di * 9 + dj] = fmaf(qwin[di], kv[dj], acc[di * 9 + dj]);

#pragma unroll
        for (int di = 8; di > 0; --di) qwin[di] = qwin[di - 1];
        qwin[0] = qnext;
    }

    const float SCALE = 0.08838834764831845f;    // 128^-0.5
#pragma unroll
    for (int ij = 0; ij < 81; ++ij)              // out[(ij*B + b)*D + d, w]
        out[(size_t)ij * (BB * DD * WW) + (size_t)bd * WW + tid] =
            acc[ij] * SCALE + rpb[ij];
}

// ---------------------------------------------------------------------------
extern "C" void kernel_launch(void* const* d_in, const int* in_sizes, int n_in,
                              void* d_out, int out_size, void* d_ws, size_t ws_size,
                              hipStream_t stream)
{
    const float* x    = (const float*)d_in[0];
    const float* Wk   = (const float*)d_in[1];
    // d_in[2] = bk : constant along the LN axis -> exactly cancelled by LN
    const float* gk   = (const float*)d_in[3];
    const float* bkln = (const float*)d_in[4];
    const float* Wq   = (const float*)d_in[5];
    // d_in[6] = bq : cancelled by LN, unused
    const float* gq   = (const float*)d_in[7];
    const float* bqln = (const float*)d_in[8];
    const float* rpb  = (const float*)d_in[9];

    const size_t plane = (size_t)BB * DD * HH * WW;   // 67,108,864 floats
    float* kbuf = (float*)d_ws;                        // 256 MB
    float* qbuf = kbuf + plane;                        // 256 MB

    conv_ln_kernel<<<BB * HH, 256, 0, stream>>>(x, Wk, gk, bkln, Wq, gq, bqln,
                                                kbuf, qbuf);
    corr_kernel<<<BB * DD, 128, 0, stream>>>(qbuf, kbuf, rpb, (float*)d_out);
}
